// RAMMultiHeadShared_27668179321267
// MI455X (gfx1250) — 8-bit matrix operand order repaired and checked by reading, compiled, not run on hardware
//
#include <hip/hip_runtime.h>


#define NUM_HEADS 8
#define IN_BITS   64
#define N_STATE   256
#define TOTAL_IN  320
#define N_OUT     64
#define K_CONN    8
#define HASH      65536
#define BATCH     128
#define TSTEPS    128

#define WAVES     16
#define THREADS   (WAVES * 32)
#define MROWS     16
#define XSTRIDE   336

typedef __attribute__((ext_vector_type(8))) int v8i;

__global__ __launch_bounds__(THREADS)
void ram_scan_wmma_kernel(const int*   __restrict__ bits,
                          const int*   __restrict__ state_coeffs,
                          const float* __restrict__ state_mem,
                          const int*   __restrict__ head_conn,
                          const int*   __restrict__ head_coeffs,
                          const float* __restrict__ head_mem,
                          float*       __restrict__ out)
{
    __shared__ unsigned char xbuf[2][MROWS][XSTRIDE];

    const int tid  = threadIdx.x;
    const int wave = tid >> 5;
    const int lane = tid & 31;
    const int hi16 = lane >> 4;
    const int l15  = lane & 15;
    const int b0   = blockIdx.x * MROWS;

    const int n_glob = wave * 16 + l15;
    v8i b_lo[5], b_hi[5];
#pragma unroll
    for (int kc = 0; kc < 5; ++kc) {
#pragma unroll
        for (int v = 0; v < 8; ++v) {
            const int kbase = 64 * kc + (v >> 2) * 32 + hi16 * 16 + (v & 3) * 4;
            unsigned int lo = 0u, hi = 0u;
#pragma unroll
            for (int by = 0; by < 4; ++by) {
                const unsigned int c =
                    (unsigned int)state_coeffs[n_glob * TOTAL_IN + kbase + by];
                lo |= (c & 0xFFu)        << (8 * by);
                hi |= ((c >> 8) & 0xFFu) << (8 * by);
            }
            b_lo[kc][v] = (int)lo;
            b_hi[kc][v] = (int)hi;
        }
    }

    const int j0 = tid, j1 = tid + THREADS;
    const int m0 = j0 >> 6, c0 = j0 & 63;
    const int m1 = j1 >> 6, c1 = j1 & 63;
    const int* const bp0 = bits + (b0 + m0) * (TSTEPS * IN_BITS) + c0;
    const int* const bp1 = bits + (b0 + m1) * (TSTEPS * IN_BITS) + c1;

    {
        unsigned char* p0 = &xbuf[0][0][0];
        for (int j = tid; j < MROWS * XSTRIDE; j += THREADS) p0[j] = 0;
    }
    __syncthreads();
    xbuf[0][m0][c0] = (unsigned char)bp0[0];
    xbuf[0][m1][c1] = (unsigned char)bp1[0];
    __syncthreads();

    const unsigned int nbase = (unsigned int)n_glob << 16;

    for (int t = 0; t < TSTEPS; ++t) {
        const int cur = t & 1, nxt = cur ^ 1;

        v8i a[5];
#pragma unroll
        for (int kc = 0; kc < 5; ++kc) {
#pragma unroll
            for (int v = 0; v < 8; ++v) {
                const int kb = 64 * kc + (v >> 2) * 32 + hi16 * 16 + (v & 3) * 4;
                a[kc][v] = *(const int*)&xbuf[cur][l15][kb];
            }
        }

        v8i acc_lo = {};
        v8i acc_hi = {};
#pragma unroll
        for (int kc = 0; kc < 5; ++kc) {
            acc_lo = __builtin_amdgcn_wmma_i32_16x16x64_iu8(
                false, a[kc], false, b_lo[kc], acc_lo, false, false);
            acc_hi = __builtin_amdgcn_wmma_i32_16x16x64_iu8(
                false, a[kc], false, b_hi[kc], acc_hi, false, false);
        }

        const int tn = (t + 1) & (TSTEPS - 1);
        const int w0 = bp0[tn * IN_BITS];
        const int w1 = bp1[tn * IN_BITS];

#pragma unroll
        for (int i = 0; i < 8; ++i) {
            const int m = i + hi16 * 8;
            const unsigned int addr =
                ((unsigned int)(acc_lo[i] + (acc_hi[i] << 8))) & 0xFFFFu;
            const float val = state_mem[nbase + addr];
            xbuf[nxt][m][IN_BITS + n_glob] = (val >= 0.5f) ? 1 : 0;
        }

        xbuf[nxt][m0][c0] = (unsigned char)w0;
        xbuf[nxt][m1][c1] = (unsigned char)w1;
        __syncthreads();
    }

    for (int item = tid; item < MROWS * N_OUT; item += THREADS) {
        const int m = item >> 6;
        const int o = item & 63;
        const int h = ((int)xbuf[1][m][61] * 4 +
                       (int)xbuf[1][m][62] * 2 +
                       (int)xbuf[1][m][63]) & (NUM_HEADS - 1);
        const int base = (h * N_OUT + o) * K_CONN;
        unsigned int acc = 0u;
#pragma unroll
        for (int k = 0; k < K_CONN; ++k) {
            const int conn = head_conn[base + k];
            const unsigned int cf = (unsigned int)head_coeffs[base + k];
            acc += (unsigned int)xbuf[0][m][IN_BITS + conn] * cf;
        }
        const unsigned int addr = acc & 0xFFFFu;
        out[(b0 + m) * N_OUT + o] =
            head_mem[(size_t)(h * N_OUT + o) * HASH + addr];
    }
}

extern "C" void kernel_launch(void* const* d_in, const int* in_sizes, int n_in,
                              void* d_out, int out_size, void* d_ws, size_t ws_size,
                              hipStream_t stream) {
    (void)in_sizes; (void)n_in; (void)out_size; (void)d_ws; (void)ws_size;
    const int*   bits         = (const int*)  d_in[0];
    const int*   state_coeffs = (const int*)  d_in[1];
    const float* state_mem    = (const float*)d_in[2];
    const int*   head_conn    = (const int*)  d_in[3];
    const int*   head_coeffs  = (const int*)  d_in[4];
    const float* head_mem     = (const float*)d_in[5];
    float*       out          = (float*)      d_out;

    ram_scan_wmma_kernel<<<BATCH / MROWS, THREADS, 0, stream>>>(
        bits, state_coeffs, state_mem, head_conn, head_coeffs, head_mem, out);
}
